// EnvLSTM_86577950753068
// MI455X (gfx1250) — matrix operand order repaired and checked by reading, not compiled, not run on hardware
//
#include <hip/hip_runtime.h>
#include <cstdint>

#define T_LEN 4096
#define DIM   1024
#define NWG   32      // workgroups per direction in the scan
#define ROWS  32      // rows of each U matrix owned per workgroup

typedef __attribute__((ext_vector_type(16))) __bf16 v16bf;
typedef __attribute__((ext_vector_type(8)))  float  v8f;

union ABFrag { uint4 u4[2]; v16bf v; };

__device__ __forceinline__ uint32_t f2bf_bits(float f) {
  uint32_t u = __float_as_uint(f);
  return (u + 0x7FFFu + ((u >> 16) & 1u)) >> 16;   // round-to-nearest-even
}
__device__ __forceinline__ float bf_lo(uint32_t u) { return __uint_as_float(u << 16); }
__device__ __forceinline__ float bf_hi(uint32_t u) { return __uint_as_float(u & 0xFFFF0000u); }

// ---------------- f32 -> bf16 conversion ----------------
__global__ __launch_bounds__(256) void k_cvt_bf16(const float* __restrict__ src,
                                                  uint16_t* __restrict__ dst, int n) {
  int i = blockIdx.x * 256 + threadIdx.x;
  if (i < n) dst[i] = (uint16_t)f2bf_bits(src[i]);
}

__global__ void k_init_bar(unsigned int* bar) {
  if (threadIdx.x < 2) bar[threadIdx.x] = 0u;
}

// ---------------- Phase 1: gx[g][t][m] = sum_i x[t][i] * W_g[m][i] ----------------
// One wave -> 16(M=t) x 64(N) tile. W concatenated as [6*DIM][DIM] bf16.
__global__ __launch_bounds__(256) void k_gemm_gx(const uint16_t* __restrict__ xbf,
                                                 const uint16_t* __restrict__ wbf,
                                                 float* __restrict__ gx) {
  const int tid  = threadIdx.x;
  const int lane = tid & 31;
  const int wid  = blockIdx.x * 8 + (tid >> 5);
  const int mt   = wid & 255;          // 256 M-tiles
  const int ntg  = wid >> 8;           // 96 N-groups of 64
  const int l16  = lane & 15;
  const int hi   = lane >> 4;
  const int row  = mt * 16 + l16;
  const int n0   = ntg * 64;

  v8f acc[4] = {};
  const uint16_t* arow = xbf + (size_t)row * DIM;
  for (int k0 = 0; k0 < DIM; k0 += 32) {
    ABFrag a;
    const uint4* ap = (const uint4*)(arow + k0 + hi * 8);
    a.u4[0] = ap[0];        // K = k0+hi*8 .. +7
    a.u4[1] = ap[2];        // K = k0+16+hi*8 .. +7
#pragma unroll
    for (int nt = 0; nt < 4; ++nt) {
      const int n = n0 + nt * 16 + l16;
      ABFrag b;
      const uint4* bp = (const uint4*)(wbf + (size_t)n * DIM + k0 + hi * 8);
      b.u4[0] = bp[0];
      b.u4[1] = bp[2];
      acc[nt] = __builtin_amdgcn_wmma_f32_16x16x32_bf16(false, a.v, false, b.v,
                                                        (short)0, acc[nt], false, false);
    }
  }
#pragma unroll
  for (int nt = 0; nt < 4; ++nt) {
    const int ngb = n0 + nt * 16;
    const int g   = ngb >> 10;                 // gate index 0..5
    const int m   = (ngb & 1023) + l16;
#pragma unroll
    for (int r = 0; r < 8; ++r) {
      const int trow = mt * 16 + hi * 8 + r;
      gx[((size_t)g * T_LEN + trow) * DIM + m] = acc[nt][r];
    }
  }
}

// ---------------- Phase 2: persistent bidirectional gated scan ----------------
struct ScanArgs {
  const float* u[6];      // uf_l, ui_l, uc_l, uf_r, ui_r, uc_r
  const float* b[6];
  const float* c0[2];
  const float* gx;        // [6][T][DIM] f32
  uint16_t*    envbf;     // [2][T][DIM] bf16 (pre-update cell states)
  float*       cbuf;      // [2 dirs][2 phases][DIM] f32
  unsigned int* bar;      // [2] monotonic arrival counters
};

__global__ __launch_bounds__(256) void k_scan(ScanArgs A) {
  __shared__ uint32_t uld[3][ROWS][513];            // bf16-pair packed U slices (padded)
  __shared__ __align__(16) float c_lds[DIM];        // full current cell state
  __shared__ float red[96][8];
  __shared__ float pa[96];

  const int tid  = threadIdx.x;
  const int dir  = blockIdx.x / NWG;
  const int wg   = blockIdx.x % NWG;
  const int row0 = wg * ROWS;

  // Prologue: stage this WG's 32-row slice of Uf/Ui/Uc into LDS as packed bf16.
  for (int idx = tid; idx < 3 * ROWS * 512; idx += 256) {
    const int gate = idx / (ROWS * 512);
    const int rem  = idx % (ROWS * 512);
    const int r    = rem / 512;
    const int p    = rem % 512;
    const float* src = A.u[dir * 3 + gate] + (size_t)(row0 + r) * DIM + p * 2;
    uld[gate][r][p] = f2bf_bits(src[0]) | (f2bf_bits(src[1]) << 16);
  }
  for (int i = tid; i < DIM; i += 256) c_lds[i] = A.c0[dir][i];
  __syncthreads();

  const int sub = tid & 7;     // 8 threads cooperate per (gate,row)
  const int grp = tid >> 3;    // 0..31 -> row
  unsigned int* barp = A.bar + dir;
  float* cb = A.cbuf + (size_t)dir * 2 * DIM;

  for (int t = 0; t < T_LEN; ++t) {
    const int tt = dir ? (T_LEN - 1 - t) : t;

    // Recurrent matvec partials: gate pass p handles gate p, row = grp.
    for (int gate = 0; gate < 3; ++gate) {
      const uint32_t* urow = uld[gate][grp];
      float s0 = 0.f, s1 = 0.f;
#pragma unroll 4
      for (int jj = 0; jj < 64; ++jj) {
        const int p = sub + (jj << 3);
        const uint32_t u = urow[p];
        const float2 c2 = *(const float2*)&c_lds[p << 1];
        s0 = fmaf(bf_lo(u), c2.x, s0);
        s1 = fmaf(bf_hi(u), c2.y, s1);
      }
      red[gate * 32 + grp][sub] = s0 + s1;
    }
    __syncthreads();

    if (tid < 96) {
      const int gate = tid >> 5, r = tid & 31, row = row0 + r;
      float s = 0.f;
#pragma unroll
      for (int j = 0; j < 8; ++j) s += red[tid][j];
      s += A.gx[((size_t)(dir * 3 + gate) * T_LEN + tt) * DIM + row];
      s += A.b[dir * 3 + gate][row];
      pa[tid] = s;
    }
    __syncthreads();

    if (tid < ROWS) {
      const int row = row0 + tid;
      const float f = 1.f / (1.f + __expf(-pa[tid]));
      const float i = 1.f / (1.f + __expf(-pa[32 + tid]));
      const float g = tanhf(pa[64 + tid]);
      const float cold = c_lds[row];
      A.envbf[((size_t)dir * T_LEN + tt) * DIM + row] = (uint16_t)f2bf_bits(cold);
      cb[(size_t)(t & 1) * DIM + row] = f * cold + i * g;   // publish owned slice
    }
    __syncthreads();

    // Device-scope barrier across the 32 WGs of this direction (monotonic counter).
    if (tid == 0) {
      __threadfence();   // release owned cbuf slice
      __hip_atomic_fetch_add(barp, 1u, __ATOMIC_RELEASE, __HIP_MEMORY_SCOPE_AGENT);
      const unsigned int target = (unsigned int)(t + 1) * NWG;
      while (__hip_atomic_load(barp, __ATOMIC_ACQUIRE, __HIP_MEMORY_SCOPE_AGENT) < target)
        __builtin_amdgcn_s_sleep(1);
    }
    __syncthreads();
    __threadfence();     // per-wave acquire: invalidate so cbuf reads are fresh

    for (int i = tid; i < DIM; i += 256)
      c_lds[i] = cb[(size_t)(t & 1) * DIM + i];
    __syncthreads();
  }
}

// ---------------- Phase 3: out = tanh(lenv @ wo^T + renv @ uo^T + bo) ----------------
__global__ __launch_bounds__(256) void k_gemm_out(const uint16_t* __restrict__ envbf,
                                                  const uint16_t* __restrict__ wobf,
                                                  const uint16_t* __restrict__ uobf,
                                                  const float* __restrict__ bo,
                                                  float* __restrict__ out) {
  const int tid  = threadIdx.x;
  const int lane = tid & 31;
  const int wid  = blockIdx.x * 8 + (tid >> 5);
  const int mt   = wid & 255;
  const int ntg  = wid >> 8;          // 0..15
  const int l16  = lane & 15;
  const int hi   = lane >> 4;
  const int row  = mt * 16 + l16;
  const int n0   = ntg * 64;

  v8f acc[4] = {};
  const uint16_t* a1row = envbf + (size_t)row * DIM;                 // lenv
  const uint16_t* a2row = envbf + ((size_t)T_LEN + row) * DIM;       // renv
  for (int k0 = 0; k0 < DIM; k0 += 32) {
    ABFrag a1, a2;
    const uint4* a1p = (const uint4*)(a1row + k0 + hi * 8);
    a1.u4[0] = a1p[0]; a1.u4[1] = a1p[2];
    const uint4* a2p = (const uint4*)(a2row + k0 + hi * 8);
    a2.u4[0] = a2p[0]; a2.u4[1] = a2p[2];
#pragma unroll
    for (int nt = 0; nt < 4; ++nt) {
      const int n = n0 + nt * 16 + l16;
      ABFrag b1, b2;
      const uint4* b1p = (const uint4*)(wobf + (size_t)n * DIM + k0 + hi * 8);
      b1.u4[0] = b1p[0]; b1.u4[1] = b1p[2];
      acc[nt] = __builtin_amdgcn_wmma_f32_16x16x32_bf16(false, a1.v, false, b1.v,
                                                        (short)0, acc[nt], false, false);
      const uint4* b2p = (const uint4*)(uobf + (size_t)n * DIM + k0 + hi * 8);
      b2.u4[0] = b2p[0]; b2.u4[1] = b2p[2];
      acc[nt] = __builtin_amdgcn_wmma_f32_16x16x32_bf16(false, a2.v, false, b2.v,
                                                        (short)0, acc[nt], false, false);
    }
  }
#pragma unroll
  for (int nt = 0; nt < 4; ++nt) {
    const int col  = n0 + nt * 16 + l16;
    const float bb = bo[col];
#pragma unroll
    for (int r = 0; r < 8; ++r) {
      const int trow = mt * 16 + hi * 8 + r;
      out[(size_t)trow * DIM + col] = tanhf(acc[nt][r] + bb);
    }
  }
}

// ---------------- host ----------------
extern "C" void kernel_launch(void* const* d_in, const int* in_sizes, int n_in,
                              void* d_out, int out_size, void* d_ws, size_t ws_size,
                              hipStream_t stream) {
  (void)in_sizes; (void)n_in; (void)out_size; (void)ws_size;
  const float* x = (const float*)d_in[0];
  const float* wmat[6] = {(const float*)d_in[1],  (const float*)d_in[4],  (const float*)d_in[7],
                          (const float*)d_in[10], (const float*)d_in[13], (const float*)d_in[16]};
  const float* umat[6] = {(const float*)d_in[2],  (const float*)d_in[5],  (const float*)d_in[8],
                          (const float*)d_in[11], (const float*)d_in[14], (const float*)d_in[17]};
  const float* bvec[6] = {(const float*)d_in[3],  (const float*)d_in[6],  (const float*)d_in[9],
                          (const float*)d_in[12], (const float*)d_in[15], (const float*)d_in[18]};
  const float* wo  = (const float*)d_in[19];
  const float* uo  = (const float*)d_in[20];
  const float* bo  = (const float*)d_in[21];
  const float* c0l = (const float*)d_in[22];
  const float* c0r = (const float*)d_in[23];

  char* ws = (char*)d_ws;
  size_t off = 0;
  uint16_t* xbf  = (uint16_t*)(ws + off); off += (size_t)T_LEN * DIM * 2;   //   8 MB
  uint16_t* wbf  = (uint16_t*)(ws + off); off += (size_t)6 * DIM * DIM * 2; //  12 MB
  uint16_t* wobf = (uint16_t*)(ws + off); off += (size_t)DIM * DIM * 2;     //   2 MB
  uint16_t* uobf = (uint16_t*)(ws + off); off += (size_t)DIM * DIM * 2;     //   2 MB
  uint16_t* envbf= (uint16_t*)(ws + off); off += (size_t)2 * T_LEN * DIM * 2; // 16 MB
  float*    gx   = (float*)   (ws + off); off += (size_t)6 * T_LEN * DIM * 4; // 96 MB
  float*    cbuf = (float*)   (ws + off); off += (size_t)2 * 2 * DIM * 4;
  unsigned int* bar = (unsigned int*)(ws + off); off += 256;

  k_cvt_bf16<<<(T_LEN * DIM + 255) / 256, 256, 0, stream>>>(x, xbf, T_LEN * DIM);
  for (int g = 0; g < 6; ++g)
    k_cvt_bf16<<<(DIM * DIM + 255) / 256, 256, 0, stream>>>(wmat[g], wbf + (size_t)g * DIM * DIM, DIM * DIM);
  k_cvt_bf16<<<(DIM * DIM + 255) / 256, 256, 0, stream>>>(wo, wobf, DIM * DIM);
  k_cvt_bf16<<<(DIM * DIM + 255) / 256, 256, 0, stream>>>(uo, uobf, DIM * DIM);
  k_init_bar<<<1, 32, 0, stream>>>(bar);

  k_gemm_gx<<<3072, 256, 0, stream>>>(xbf, wbf, gx);   // 24576 waves: 256 Mtiles x 96 Ngroups

  ScanArgs A;
  for (int g = 0; g < 6; ++g) { A.u[g] = umat[g]; A.b[g] = bvec[g]; }
  A.c0[0] = c0l; A.c0[1] = c0r;
  A.gx = gx; A.envbf = envbf; A.cbuf = cbuf; A.bar = bar;
  k_scan<<<2 * NWG, 256, 0, stream>>>(A);              // 64 persistent WGs (1/WGP)

  k_gemm_out<<<512, 256, 0, stream>>>(envbf, wobf, uobf, bo, (float*)d_out);
}
